// MultiHeadPCAProjection_15625091023504
// MI455X (gfx1250) — matrix operand order repaired and checked by reading, not compiled, not run on hardware
//
#include <hip/hip_runtime.h>

typedef __attribute__((ext_vector_type(16))) __bf16     v16bf;
typedef __attribute__((ext_vector_type(2)))  __bf16     v2bf;
typedef __attribute__((ext_vector_type(8)))  float      v8f;
typedef __attribute__((ext_vector_type(8)))  unsigned   v8u;

#define B_  8
#define S_  4096
#define E_  1024
#define H_  8
#define HS_ 128
#define M_  (B_*S_)   // 32768

// LDS row pitch for packed-bf16 tiles: 20 dwords = 80B.
//  - keeps 16B alignment (ds_load_b128 merging legal)
//  - 20*r mod 64 hits 16 distinct banks for r=0..15 -> conflict-free fragments
#define PITCH 20

// Packed f32->bf16 pair; v2bf build lets the backend emit v_cvt_pk_bf16_f32.
static __device__ __forceinline__ unsigned pack_bf16(float a, float b) {
    v2bf p;
    p[0] = (__bf16)a;
    p[1] = (__bf16)b;
    return __builtin_bit_cast(unsigned, p);
}

// ---------------------------------------------------------------------------
// Kernel 1: yc = center_over_head( relu(x @ W + b) )
// grid (M_/128, E_/128) = (256, 8); block = 256 threads (8 waves).
// LDS tiles stored as bf16 pairs packed along K: fragments are contiguous
// dword runs per lane (b128-mergeable), conversion done once at staging.
// ---------------------------------------------------------------------------
__global__ __launch_bounds__(256)
void k1_dense_relu_center(const float* __restrict__ x,
                          const float* __restrict__ W,
                          const float* __restrict__ bias,
                          float* __restrict__ yc)
{
    __shared__ unsigned Xt[128][PITCH];  // [m][k-pair] packed bf16
    __shared__ unsigned Wt[128][PITCH];  // [n][k-pair] packed bf16 (transposed)
    __shared__ float    Ot[128][129];    // f32 output tile for centering

    const int m0   = blockIdx.x * 128;
    const int n0   = blockIdx.y * 128;
    const int wave = threadIdx.x >> 5;
    const int lane = threadIdx.x & 31;
    const int hi   = lane >> 4;
    const int ln   = lane & 15;

    v8f acc[8];
    const v8f vzero = {};
    #pragma unroll
    for (int t = 0; t < 8; ++t) acc[t] = vzero;

    for (int k0 = 0; k0 < E_; k0 += 32) {
        __syncthreads();
        // stage X: 128 rows x 16 k-pairs; float2 load, pack, 1 dword store
        #pragma unroll
        for (int i = 0; i < 8; ++i) {
            int idx = threadIdx.x + 256 * i;           // 0..2047
            int row = idx >> 4, kp = idx & 15;
            const float* p = x + (size_t)(m0 + row) * E_ + k0 + 2 * kp;
            Xt[row][kp] = pack_bf16(p[0], p[1]);
        }
        // stage W transposed: lanes sweep n (coalesced), two k-rows per pair
        #pragma unroll
        for (int i = 0; i < 8; ++i) {
            int idx = threadIdx.x + 256 * i;           // 0..2047
            int n = idx & 127, kp = idx >> 7;          // kp 0..15
            const float* p = W + (size_t)(k0 + 2 * kp) * E_ + n0 + n;
            Wt[n][kp] = pack_bf16(p[0], p[E_]);
        }
        __syncthreads();

        // A fragment: two contiguous 4-dword runs (K = 8*hi..+7, 16+8*hi..+7)
        v8u au;
        const unsigned* xr = Xt[wave * 16 + ln];
        #pragma unroll
        for (int t = 0; t < 4; ++t) au[t]     = xr[4 * hi + t];
        #pragma unroll
        for (int t = 0; t < 4; ++t) au[4 + t] = xr[8 + 4 * hi + t];
        const v16bf a = __builtin_bit_cast(v16bf, au);

        #pragma unroll
        for (int jt = 0; jt < 8; ++jt) {
            // B fragment: two contiguous 4-dword runs (K = 8*hi..+7, 16+8*hi..+7)
            v8u bu;
            const unsigned* wr = Wt[jt * 16 + ln];
            #pragma unroll
            for (int t = 0; t < 4; ++t) bu[t]     = wr[4 * hi + t];
            #pragma unroll
            for (int t = 0; t < 4; ++t) bu[4 + t] = wr[8 + 4 * hi + t];
            acc[jt] = __builtin_amdgcn_wmma_f32_16x16x32_bf16(
                false, a, false, __builtin_bit_cast(v16bf, bu),
                (short)0, acc[jt], false, false);
        }
    }

    // epilogue: bias + relu into LDS
    __syncthreads();
    #pragma unroll
    for (int jt = 0; jt < 8; ++jt) {
        const float bv = bias[n0 + jt * 16 + ln];
        #pragma unroll
        for (int r = 0; r < 8; ++r) {
            float v = acc[jt][r] + bv;
            Ot[wave * 16 + r + 8 * hi][jt * 16 + ln] = v > 0.f ? v : 0.f;
        }
    }
    __syncthreads();
    // mean-center across the head feature axis (this 128-col block = one head)
    if (threadIdx.x < 128) {
        const int r = threadIdx.x;
        float s = 0.f;
        #pragma unroll 8
        for (int j = 0; j < 128; ++j) s += Ot[r][j];
        const float mean = s * (1.f / 128.f);
        float* dst = yc + (size_t)(m0 + r) * E_ + n0;
        for (int j = 0; j < 128; ++j) dst[j] = Ot[r][j] - mean;
    }
}

// ---------------------------------------------------------------------------
// Kernel 2: partial cov[b,h] = Xc^T Xc over an S-chunk (split-K for occupancy:
// 64 x nchunks workgroups instead of 64). grid (64, nchunks); block 256.
// One packed-bf16 tile T[i][s-pair] feeds both A and B fragments.
// ---------------------------------------------------------------------------
__global__ __launch_bounds__(256)
void k2_cov(const float* __restrict__ yc, float* __restrict__ covp, int schunk)
{
    __shared__ unsigned T[128][PITCH];   // [i][s-pair] packed bf16

    const int bh = blockIdx.x;
    const int c  = blockIdx.y;
    const int b  = bh >> 3, h = bh & 7;
    const float* base = yc + (size_t)b * S_ * E_ + (size_t)h * HS_;
    const int wave = threadIdx.x >> 5;
    const int lane = threadIdx.x & 31;
    const int hi   = lane >> 4;
    const int ln   = lane & 15;
    const int i0   = wave * 16;
    const int s_begin = c * schunk;

    v8f acc[8];
    const v8f vzero = {};
    #pragma unroll
    for (int t = 0; t < 8; ++t) acc[t] = vzero;

    for (int s0 = s_begin; s0 < s_begin + schunk; s0 += 32) {
        __syncthreads();
        // stage transposed: lanes sweep i (coalesced), two s-rows per pair
        #pragma unroll
        for (int i = 0; i < 8; ++i) {
            int idx = threadIdx.x + 256 * i;           // 0..2047
            int ii = idx & 127, sp = idx >> 7;         // sp 0..15
            const float* p = base + (size_t)(s0 + 2 * sp) * E_ + ii;
            T[ii][sp] = pack_bf16(p[0], p[E_]);
        }
        __syncthreads();

        // A = Xc^T fragment: rows i, K = s (two 4-dword runs)
        v8u au;
        const unsigned* ar = T[i0 + ln];
        #pragma unroll
        for (int t = 0; t < 4; ++t) au[t]     = ar[4 * hi + t];
        #pragma unroll
        for (int t = 0; t < 4; ++t) au[4 + t] = ar[8 + 4 * hi + t];
        const v16bf a = __builtin_bit_cast(v16bf, au);

        #pragma unroll
        for (int jt = 0; jt < 8; ++jt) {
            v8u bu;                                    // B = Xc: cols j, K = s
            const unsigned* br = T[jt * 16 + ln];
            #pragma unroll
            for (int t = 0; t < 4; ++t) bu[t]     = br[4 * hi + t];
            #pragma unroll
            for (int t = 0; t < 4; ++t) bu[4 + t] = br[8 + 4 * hi + t];
            acc[jt] = __builtin_amdgcn_wmma_f32_16x16x32_bf16(
                false, a, false, __builtin_bit_cast(v16bf, bu),
                (short)0, acc[jt], false, false);
        }
    }

    float* C = covp + ((size_t)bh * gridDim.y + c) * HS_ * HS_;
    #pragma unroll
    for (int jt = 0; jt < 8; ++jt)
        #pragma unroll
        for (int r = 0; r < 8; ++r)
            C[(size_t)(i0 + r + 8 * hi) * HS_ + jt * 16 + ln] = acc[jt][r];
}

// Deterministic split-K reduction: cov = sum over chunks of covp.
__global__ __launch_bounds__(256)
void k2r_reduce(const float* __restrict__ covp, float* __restrict__ cov,
                int nchunks)
{
    const size_t idx = (size_t)blockIdx.x * 256 + threadIdx.x; // 64*16384 total
    const size_t bh  = idx >> 14;
    const size_t e   = idx & 16383;
    float s = 0.f;
    for (int c = 0; c < nchunks; ++c)
        s += covp[(bh * (size_t)nchunks + c) * (HS_ * HS_) + e];
    cov[idx] = s;
}

// ---------------------------------------------------------------------------
// Kernel 3: parallel two-sided Jacobi eigensolver (128x128, chess-tournament
// ordering, 64 disjoint pairs/round) + rank-sorted projection onto u.
// One workgroup (256 threads) per (b,h); A and V resident in 320KB WGP LDS.
// ---------------------------------------------------------------------------
__global__ __launch_bounds__(256)
void k3_jacobi_project(const float* __restrict__ cov,
                       const float* __restrict__ u,
                       float* __restrict__ out)
{
    __shared__ float A[128][129];
    __shared__ float V[128][129];
    __shared__ float rc[64], rs[64];
    __shared__ int   pp[64], qq[64];
    __shared__ int   top[64], bot[64], tn[64], bn[64];
    __shared__ float ev[128], up[128];

    const int bh  = blockIdx.x;
    const int tid = threadIdx.x;
    const float* C = cov + (size_t)bh * 128 * 128;

    for (int idx = tid; idx < 128 * 128; idx += 256) {
        int i = idx >> 7, j = idx & 127;
        A[i][j] = C[idx];
        V[i][j] = (i == j) ? 1.f : 0.f;
    }
    if (tid < 64) { top[tid] = 2 * tid; bot[tid] = 2 * tid + 1; }
    __syncthreads();

    for (int sweep = 0; sweep < 8; ++sweep) {
        for (int round = 0; round < 127; ++round) {
            // phase 1: 64 rotation angles + next tournament permutation
            if (tid < 64) {
                int p = top[tid], q = bot[tid];
                if (p > q) { int t = p; p = q; q = t; }
                pp[tid] = p; qq[tid] = q;
                float apq = A[p][q];
                float c = 1.f, s = 0.f;
                if (fabsf(apq) > 1e-12f) {
                    float tau = (A[q][q] - A[p][p]) / (2.f * apq);
                    float t = (tau >= 0.f)
                        ? 1.f / (tau + sqrtf(1.f + tau * tau))
                        : 1.f / (tau - sqrtf(1.f + tau * tau));
                    c = 1.f / sqrtf(1.f + t * t);
                    s = t * c;
                }
                rc[tid] = c; rs[tid] = s;
                int nt = (tid == 0) ? top[0] : (tid == 1 ? bot[0] : top[tid - 1]);
                int nb = (tid == 63) ? top[63] : bot[tid + 1];
                tn[tid] = nt; bn[tid] = nb;
            }
            __syncthreads();
            // phase 2: row rotations (disjoint row pairs -> race-free)
            for (int idx = tid; idx < 64 * 128; idx += 256) {
                int k = idx >> 7, j = idx & 127;
                int p = pp[k], q = qq[k];
                float c = rc[k], s = rs[k];
                float ap = A[p][j], aq = A[q][j];
                A[p][j] = c * ap - s * aq;
                A[q][j] = s * ap + c * aq;
            }
            __syncthreads();
            // phase 3: column rotations on A and V (disjoint column pairs)
            for (int idx = tid; idx < 64 * 128; idx += 256) {
                int k = idx >> 7, i = idx & 127;
                int p = pp[k], q = qq[k];
                float c = rc[k], s = rs[k];
                float ap = A[i][p], aq = A[i][q];
                A[i][p] = c * ap - s * aq;
                A[i][q] = s * ap + c * aq;
                float vp = V[i][p], vq = V[i][q];
                V[i][p] = c * vp - s * vq;
                V[i][q] = s * vp + c * vq;
            }
            __syncthreads();
            if (tid < 64) { top[tid] = tn[tid]; bot[tid] = bn[tid]; }
            __syncthreads();
        }
    }

    // eigenvalue ranks (ascending, eigh order) -> gather u[rank]
    if (tid < 128) ev[tid] = A[tid][tid];
    __syncthreads();
    if (tid < 128) {
        float lj = ev[tid];
        int r = 0;
        for (int k = 0; k < 128; ++k) {
            float lk = ev[k];
            r += (lk < lj || (lk == lj && k < tid)) ? 1 : 0;
        }
        up[tid] = u[r];
    }
    __syncthreads();
    // projs[i] = sum_j V[i][j] * u[rank(j)] ; out[b, i*H + h]
    if (tid < 128) {
        float s = 0.f;
        for (int j = 0; j < 128; ++j) s += V[tid][j] * up[j];
        int b = bh >> 3, h = bh & 7;
        out[(size_t)b * E_ + tid * H_ + h] = s;
    }
}

// ---------------------------------------------------------------------------
extern "C" void kernel_launch(void* const* d_in, const int* in_sizes, int n_in,
                              void* d_out, int out_size, void* d_ws, size_t ws_size,
                              hipStream_t stream) {
    const float* x    = (const float*)d_in[0];  // [B,S,E]
    const float* W    = (const float*)d_in[1];  // [E,E]
    const float* bias = (const float*)d_in[2];  // [E]
    const float* u    = (const float*)d_in[3];  // [1,HS]
    float* out = (float*)d_out;                 // [B,E]

    const size_t yc_elems  = (size_t)M_ * E_;           // 33,554,432
    const size_t cov_elems = (size_t)B_ * H_ * HS_ * HS_; // 1,048,576

    float* yc = (float*)d_ws;                            // 128 MB

    // split-K covariance if scratch allows (ws_size is a launch constant ->
    // the chosen path is identical on every call: deterministic)
    const int    NCHUNK = 8;
    const size_t need_split = (yc_elems + cov_elems * (NCHUNK + 1)) * sizeof(float);
    const int nchunks = (ws_size >= need_split) ? NCHUNK : 1;

    float* covp = yc + yc_elems;                               // partials
    float* cov  = (nchunks > 1) ? covp + cov_elems * nchunks   // reduced
                                : covp;

    k1_dense_relu_center<<<dim3(M_ / 128, E_ / 128), dim3(256), 0, stream>>>(x, W, bias, yc);
    k2_cov<<<dim3(B_ * H_, nchunks), dim3(256), 0, stream>>>(yc, covp, S_ / nchunks);
    if (nchunks > 1)
        k2r_reduce<<<dim3((unsigned)(cov_elems / 256)), dim3(256), 0, stream>>>(covp, cov, nchunks);
    k3_jacobi_project<<<dim3(B_ * H_), dim3(256), 0, stream>>>(cov, u, out);
}
